// Transformer_13451837571781
// MI455X (gfx1250) — matrix operand order repaired and checked by reading, not compiled, not run on hardware
//
#include <hip/hip_runtime.h>
#include <hip/hip_bf16.h>

// ---------------------------------------------------------------------------
// Model constants (from reference)
// ---------------------------------------------------------------------------
#define BSZ   8
#define SP    512
#define SR    101
#define DM    512
#define HEADS 8
#define DEPTH 64
#define DFF   2048

typedef unsigned short u16;
typedef __attribute__((ext_vector_type(16))) __bf16 v16bf;
typedef __attribute__((ext_vector_type(8)))  float  v8f;
typedef __attribute__((ext_vector_type(4)))  unsigned int v4u;

union FragU { v16bf v; v4u q[2]; };
union V4H  { v4u q; u16 h[8]; };

__device__ __forceinline__ u16 f2bf(float f) {
  unsigned u = __float_as_uint(f);
  unsigned r = (u + 0x7FFFu + ((u >> 16) & 1u)) >> 16;   // round-to-nearest-even
  return (u16)r;
}

// ---------------------------------------------------------------------------
// Tiled bf16 WMMA GEMM:  C[M,N] = epilogue( A * B + bias )
//   A: bf16, row-major MxK (TA=0) or KxM (TA=1, i.e. A[m][k]=Asrc[k*lda+m])
//   B: bf16, row-major KxN (TB=0) or NxK (TB=1, i.e. B[k][n]=Bsrc[n*ldb+k])
//   C: optional f32 and/or bf16 outputs. Batched via (b,h) strides, z=b*H+h.
//   Block tile 128x64, K-tile 64 (two 16x16x32 WMMA sub-steps per stage).
//   EPI=1: fused attention-logits epilogue (aug table, additive mask, 1/sqrt(d))
//   applied in-register before the store -- avoids a full extra read+write
//   pass over the 67MB logits tensor (the dominant DRAM traffic item).
// ---------------------------------------------------------------------------
template<int TA, int TB, int ACT, int EPI>
__global__ __launch_bounds__(256)
void gemm_bf16_wmma(const u16* __restrict__ Ag, const u16* __restrict__ Bg,
                    float* __restrict__ Cf, u16* __restrict__ Cb,
                    const float* __restrict__ bias,
                    int M, int N, int K, int lda, int ldb, int ldc,
                    long sAb, long sAh, long sBb, long sBh, long sCb, long sCh,
                    int H,
                    const float* __restrict__ ehb, const float* __restrict__ epi,
                    const float* __restrict__ emask,
                    const float* __restrict__ eat, const float* __restrict__ eaug,
                    int etransp) {
  const int BM = 128, BN = 64, BKT = 64, LDAS = 72, LDBS = 72;
  __shared__ u16 sA[BM * LDAS];   // [m][k], padded rows (144B, 16B-aligned)
  __shared__ u16 sB[BN * LDBS];   // [n][k] (K-contiguous per column)

  int bz = blockIdx.z;
  int bb = bz / H, hh = bz % H;
  Ag += (long)bb * sAb + (long)hh * sAh;
  Bg += (long)bb * sBb + (long)hh * sBh;
  long cOff = (long)bb * sCb + (long)hh * sCh;

  int m0 = blockIdx.y * BM;
  int n0 = blockIdx.x * BN;
  int tid = threadIdx.x;
  int lane = tid & 31, wid = tid >> 5;
  int wm = (wid >> 1) * 32;       // 0,32,64,96
  int wn = (wid & 1) * 32;        // 0,32
  int r = lane & 15, hgrp = lane >> 4;

  const bool aAligned = ((lda & 7) == 0);
  const bool bAligned = ((ldb & 7) == 0);
  const bool mFull = (m0 + BM <= M);
  const bool nFull = (n0 + BN <= N);

  v8f acc[2][2];
  #pragma unroll
  for (int i = 0; i < 2; ++i)
    #pragma unroll
    for (int j = 0; j < 2; ++j) acc[i][j] = (v8f)0.f;

  for (int k0 = 0; k0 < K; k0 += BKT) {
    const bool kFull = (k0 + BKT <= K);

    // ------------------------- stage A tile (128x64) -----------------------
    if (!TA && kFull && mFull && aAligned) {
      // contiguous K: b128 global loads -> b128 LDS stores
      #pragma unroll
      for (int j = 0; j < 4; ++j) {
        int c = tid + j * 256;            // 1024 chunks of 8 halfs
        int row = c >> 3;
        int kc = (c & 7) << 3;
        v4u d = *(const v4u*)(Ag + (long)(m0 + row) * lda + k0 + kc);
        *(v4u*)(&sA[row * LDAS + kc]) = d;
      }
    } else if (TA && kFull && mFull && aAligned) {
      // contiguous M: b128 load along m, scatter 8 b16 stores across LDS rows
      #pragma unroll
      for (int j = 0; j < 4; ++j) {
        int c = tid + j * 256;            // 64 k x 16 m-chunks
        int kk = c >> 4;
        int mc = (c & 15) << 3;
        V4H d;
        d.q = *(const v4u*)(Ag + (long)(k0 + kk) * lda + m0 + mc);
        #pragma unroll
        for (int e = 0; e < 8; ++e) sA[(mc + e) * LDAS + kk] = d.h[e];
      }
    } else {
      // fallback: clamped unconditional loads + select (edge tiles or unaligned lda)
      #pragma unroll 1
      for (int j = 0; j < 32; ++j) {
        int s = tid + j * 256;            // 8192 elements
        int mi = s >> 6;
        int kk = s & 63;
        int gm = m0 + mi, gk = k0 + kk;
        bool inb = (gm < M) && (gk < K);
        int cm = gm < M ? gm : M - 1;
        int ck = gk < K ? gk : K - 1;
        long idx = TA ? ((long)ck * lda + cm) : ((long)cm * lda + ck);
        u16 v = Ag[idx];
        sA[mi * LDAS + kk] = inb ? v : (u16)0;
      }
    }

    // ------------------------- stage B tile (64x64 as [n][k]) --------------
    if (TB && kFull && nFull && bAligned) {
      #pragma unroll
      for (int j = 0; j < 2; ++j) {
        int c = tid + j * 256;            // 64 n x 8 k-chunks
        int ni = c >> 3;
        int kc = (c & 7) << 3;
        v4u d = *(const v4u*)(Bg + (long)(n0 + ni) * ldb + k0 + kc);
        *(v4u*)(&sB[ni * LDBS + kc]) = d;
      }
    } else if (!TB && kFull && nFull && bAligned) {
      #pragma unroll
      for (int j = 0; j < 2; ++j) {
        int c = tid + j * 256;            // 64 k x 8 n-chunks
        int kk = c >> 3;
        int nc = (c & 7) << 3;
        V4H d;
        d.q = *(const v4u*)(Bg + (long)(k0 + kk) * ldb + n0 + nc);
        #pragma unroll
        for (int e = 0; e < 8; ++e) sB[(nc + e) * LDBS + kk] = d.h[e];
      }
    } else {
      #pragma unroll 1
      for (int j = 0; j < 16; ++j) {
        int s = tid + j * 256;            // 4096 elements
        int ni = s >> 6;
        int kk = s & 63;
        int gn = n0 + ni, gk = k0 + kk;
        bool inb = (gn < N) && (gk < K);
        int cn = gn < N ? gn : N - 1;
        int ck = gk < K ? gk : K - 1;
        long idx = TB ? ((long)cn * ldb + ck) : ((long)ck * ldb + cn);
        u16 v = Bg[idx];
        sB[ni * LDBS + kk] = inb ? v : (u16)0;
      }
    }
    __syncthreads();

    // ---- two K=32 WMMA sub-steps per staged tile ----
    #pragma unroll
    for (int ks = 0; ks < 2; ++ks) {
      const int kb = ks * 32;
      // A 16x32: lane r = row; lanes<16 K={0..7,16..23}, lanes>=16 K={8..15,24..31}
      FragU a[2], b[2];
      #pragma unroll
      for (int i = 0; i < 2; ++i) {
        const u16* p = &sA[(wm + i * 16 + r) * LDAS + kb];
        a[i].q[0] = *(const v4u*)(p + hgrp * 8);
        a[i].q[1] = *(const v4u*)(p + 16 + hgrp * 8);
      }
      // B 32x16: lane r = col; lanes<16 K={0..7,16..23}, lanes>=16 K={8..15,24..31}
      #pragma unroll
      for (int j = 0; j < 2; ++j) {
        const u16* p = &sB[(wn + j * 16 + r) * LDBS + kb];
        b[j].q[0] = *(const v4u*)(p + hgrp * 8);
        b[j].q[1] = *(const v4u*)(p + 16 + hgrp * 8);
      }
      #pragma unroll
      for (int i = 0; i < 2; ++i)
        #pragma unroll
        for (int j = 0; j < 2; ++j)
          acc[i][j] = __builtin_amdgcn_wmma_f32_16x16x32_bf16(
              false, a[i].v, false, b[j].v, (short)0, acc[i][j], false, false);
    }
    __syncthreads();
  }

  // ---- epilogue: bias (+ReLU) or fused logits transform; f32/bf16 stores --
  #pragma unroll
  for (int i = 0; i < 2; ++i) {
    #pragma unroll
    for (int j = 0; j < 2; ++j) {
      int gn = n0 + wn + j * 16 + (lane & 15);
      if (gn >= N) continue;
      float bv = (!EPI && bias) ? bias[gn] : 0.f;
      #pragma unroll
      for (int v = 0; v < 8; ++v) {
        int gm = m0 + wm + i * 16 + v + 8 * (lane >> 4);
        if (gm < M) {
          float val = acc[i][j][v] + bv;
          if (ACT) val = fmaxf(val, 0.f);
          if (EPI) {
            // logits epilogue: (l*at + aug*tab[k,q]) + mask[k,q]*NEG, * 1/8
            long mi = (long)bb * M * N +
                      (etransp ? ((long)gn * M + gm) : ((long)gm * N + gn));
            if (ehb) {
              const float* tabs = (hh < HEADS / 2) ? ehb : epi;
              val = val * (*eat) + (*eaug) * tabs[mi];
            }
            if (emask) val += emask[mi] * (-1e9f);
            val *= 0.125f;                       // 1/sqrt(DEPTH)
          }
          long off = cOff + (long)gm * ldc + gn;
          if (Cf) Cf[off] = val;
          if (Cb) Cb[off] = f2bf(val);
        }
      }
    }
  }
}

// ---------------------------------------------------------------------------
// f32 -> bf16 conversion (weights)
// ---------------------------------------------------------------------------
__global__ void cvt_kernel(const float* __restrict__ in, u16* __restrict__ out, long n) {
  long i = (long)blockIdx.x * 256 + threadIdx.x;
  if (i < n) out[i] = f2bf(in[i]);
}

// ---------------------------------------------------------------------------
// Embedding: x = (table[tok]*sqrt(d) + pos_enc) * sqrt(d) / d
// ---------------------------------------------------------------------------
__global__ void embed_kernel(const int* __restrict__ tok, const float* __restrict__ table,
                             float* __restrict__ of, u16* __restrict__ ob,
                             int S, long total) {
  long i = (long)blockIdx.x * 256 + threadIdx.x;
  if (i >= total) return;
  int d = i % DM;
  long r = i / DM;
  int s = r % S;
  int b = r / S;
  float den = __powf(10000.f, (float)(d & ~1) / (float)DM);
  float ang = (float)s / den;
  float pe = (d & 1) ? __cosf(ang) : __sinf(ang);
  const float sq = 22.62741699796952f;           // sqrt(512)
  float x = (table[(long)tok[(long)b * S + s] * DM + d] * sq + pe) * (sq / (float)DM);
  of[i] = x;
  ob[i] = f2bf(x);
}

// ---------------------------------------------------------------------------
// Joint 2D softmax over (k,q) per (b,h), rescaled by row_mult; bf16 output
// ---------------------------------------------------------------------------
__global__ void softmax2d_kernel(const float* __restrict__ logits, u16* __restrict__ attn,
                                 long n, const float* __restrict__ rm_ptr, float rm_const) {
  __shared__ float red[256];
  long base = (long)blockIdx.x * n;
  const float* L = logits + base;
  int t = threadIdx.x;
  float mx = -3.4e38f;
  for (long i = t; i < n; i += 256) mx = fmaxf(mx, L[i]);
  red[t] = mx; __syncthreads();
  for (int s = 128; s > 0; s >>= 1) { if (t < s) red[t] = fmaxf(red[t], red[t + s]); __syncthreads(); }
  mx = red[0]; __syncthreads();
  float sm = 0.f;
  for (long i = t; i < n; i += 256) sm += __expf(L[i] - mx);
  red[t] = sm; __syncthreads();
  for (int s = 128; s > 0; s >>= 1) { if (t < s) red[t] += red[t + s]; __syncthreads(); }
  float rm = rm_ptr ? *rm_ptr : rm_const;
  float inv = rm / red[0];
  u16* A = attn + base;
  for (long i = t; i < n; i += 256) A[i] = f2bf(__expf(L[i] - mx) * inv);
}

// ---------------------------------------------------------------------------
// out = LayerNorm(x + a) * g + b  (D fixed 512), writes f32 and bf16
// ---------------------------------------------------------------------------
__global__ __launch_bounds__(256)
void add_ln_kernel(const float* __restrict__ x, const float* __restrict__ a,
                   const float* __restrict__ g, const float* __restrict__ bt,
                   float* __restrict__ of, u16* __restrict__ ob) {
  __shared__ float red[256];
  long row = blockIdx.x;
  int t = threadIdx.x;
  const float* xr = x + row * DM;
  const float* ar = a + row * DM;
  float v0 = xr[t] + ar[t];
  float v1 = xr[t + 256] + ar[t + 256];
  red[t] = v0 + v1; __syncthreads();
  for (int s = 128; s > 0; s >>= 1) { if (t < s) red[t] += red[t + s]; __syncthreads(); }
  float mu = red[0] * (1.f / DM); __syncthreads();
  float d0 = v0 - mu, d1 = v1 - mu;
  red[t] = d0 * d0 + d1 * d1; __syncthreads();
  for (int s = 128; s > 0; s >>= 1) { if (t < s) red[t] += red[t + s]; __syncthreads(); }
  float inv = rsqrtf(red[0] * (1.f / DM) + 1e-9f);
  float o0 = g[t] * d0 * inv + bt[t];
  float o1 = g[t + 256] * d1 * inv + bt[t + 256];
  long base = row * DM;
  of[base + t] = o0;        of[base + t + 256] = o1;
  ob[base + t] = f2bf(o0);  ob[base + t + 256] = f2bf(o1);
}

// ---------------------------------------------------------------------------
// pro_cnt = count_nonzero(protok[0, :])
// ---------------------------------------------------------------------------
__global__ void count_nz_kernel(const int* __restrict__ tok, float* __restrict__ out, int n) {
  __shared__ int red[256];
  int t = threadIdx.x, c = 0;
  for (int i = t; i < n; i += 256) c += (tok[i] != 0);
  red[t] = c; __syncthreads();
  for (int s = 128; s > 0; s >>= 1) { if (t < s) red[t] += red[t + s]; __syncthreads(); }
  if (t == 0) *out = (float)red[0];
}

// ---------------------------------------------------------------------------
// x[b,d] = 0.5*(mean_s pro + mean_s rna)
// ---------------------------------------------------------------------------
__global__ void meanpool_kernel(const float* __restrict__ pro, const float* __restrict__ rna,
                                float* __restrict__ x) {
  int i = blockIdx.x * 256 + threadIdx.x;
  if (i >= BSZ * DM) return;
  int b = i / DM, d = i % DM;
  const float* p = pro + (long)b * SP * DM + d;
  float sp = 0.f;
  for (int s = 0; s < SP; ++s) sp += p[(long)s * DM];
  const float* rr = rna + (long)b * SR * DM + d;
  float sr = 0.f;
  for (int s = 0; s < SR; ++s) sr += rr[(long)s * DM];
  x[i] = 0.5f * (sp * (1.f / SP) + sr * (1.f / SR));
}

// ---------------------------------------------------------------------------
// final head: softmax(x @ fin_w + fin_b), 2 classes
// ---------------------------------------------------------------------------
__global__ void head_kernel(const float* __restrict__ x, const float* __restrict__ w,
                            const float* __restrict__ bias, float* __restrict__ out) {
  __shared__ float r0[128], r1[128];
  int b = blockIdx.x, t = threadIdx.x;
  const float* xr = x + (long)b * DM;
  float a0 = 0.f, a1 = 0.f;
  for (int d = t; d < DM; d += 128) {
    float xv = xr[d];
    a0 += xv * w[d * 2];
    a1 += xv * w[d * 2 + 1];
  }
  r0[t] = a0; r1[t] = a1; __syncthreads();
  for (int s = 64; s > 0; s >>= 1) {
    if (t < s) { r0[t] += r0[t + s]; r1[t] += r1[t + s]; }
    __syncthreads();
  }
  if (t == 0) {
    float l0 = r0[0] + bias[0], l1 = r1[0] + bias[1];
    float m = fmaxf(l0, l1);
    float e0 = __expf(l0 - m), e1 = __expf(l1 - m);
    out[b * 2]     = e0 / (e0 + e1);
    out[b * 2 + 1] = e1 / (e0 + e1);
  }
}

// ===========================================================================
// Host orchestration
// ===========================================================================
struct AttnP {
  const float *bq, *bk, *bv, *bo, *fb1, *fb2, *g1, *b1, *g2, *b2;
  u16 *wq, *wk, *wv, *wo, *f1, *f2;
  int dff;
};

extern "C" void kernel_launch(void* const* d_in, const int* in_sizes, int n_in,
                              void* d_out, int out_size, void* d_ws, size_t ws_size,
                              hipStream_t stream) {
  (void)in_sizes; (void)n_in; (void)out_size; (void)ws_size;
  // --- flattened input indices (dict order of setup_inputs) ---
  enum { I_PRO_EMB = 0, I_RNA_EMB = 1, I_PRO_SELF = 2, I_RNA_SELF = 18,
         I_RNA_CROSS = 34, I_PRO_CROSS = 50, I_FIN_W = 66, I_FIN_B = 67,
         I_AUG_AT = 68, I_AUG_AUG = 69, I_PROTOK = 70, I_RNATOK = 71,
         I_SELF_MASK = 72, I_CROSS_MASK = 73, I_HB = 74, I_PI = 75 };
  auto F = [&](int i) { return (const float*)d_in[i]; };

  // --- bump allocator over workspace ---
  char* ws = (char*)d_ws;
  size_t off = 0;
  auto alloc = [&](size_t bytes) -> void* {
    void* p = ws + off;
    off = (off + bytes + 255) & ~(size_t)255;
    return p;
  };

  auto cvt = [&](const float* in, u16* out, long n) {
    cvt_kernel<<<dim3((unsigned)((n + 255) / 256)), dim3(256), 0, stream>>>(in, out, n);
  };

  // --- convert all GEMM weights to bf16 once per call ---
  auto mkParams = [&](int base, int dff) -> AttnP {
    AttnP p;
    p.bq = F(base + 1);  p.bk = F(base + 3);  p.bv = F(base + 5);  p.bo = F(base + 7);
    p.fb1 = F(base + 9); p.fb2 = F(base + 11);
    p.g1 = F(base + 12); p.b1 = F(base + 13); p.g2 = F(base + 14); p.b2 = F(base + 15);
    p.dff = dff;
    p.wq = (u16*)alloc((size_t)DM * DM * 2);   cvt(F(base + 0),  p.wq, (long)DM * DM);
    p.wk = (u16*)alloc((size_t)DM * DM * 2);   cvt(F(base + 2),  p.wk, (long)DM * DM);
    p.wv = (u16*)alloc((size_t)DM * DM * 2);   cvt(F(base + 4),  p.wv, (long)DM * DM);
    p.wo = (u16*)alloc((size_t)DM * DM * 2);   cvt(F(base + 6),  p.wo, (long)DM * DM);
    p.f1 = (u16*)alloc((size_t)DM * dff * 2);  cvt(F(base + 8),  p.f1, (long)DM * dff);
    p.f2 = (u16*)alloc((size_t)dff * DM * 2);  cvt(F(base + 10), p.f2, (long)dff * DM);
    return p;
  };
  AttnP pro_self  = mkParams(I_PRO_SELF,  DFF);
  AttnP rna_self  = mkParams(I_RNA_SELF,  DFF);
  AttnP rna_cross = mkParams(I_RNA_CROSS, DFF);
  AttnP pro_cross = mkParams(I_PRO_CROSS, DFF);

  // --- activation / scratch buffers ---
  const long maxTok = (long)BSZ * SP;            // 4096 rows max
  float* pro_f[2]; u16* pro_b[2];
  float* rna_f[2]; u16* rna_b[2];
  for (int i = 0; i < 2; ++i) {
    pro_f[i] = (float*)alloc((size_t)BSZ * SP * DM * 4);
    pro_b[i] = (u16*)  alloc((size_t)BSZ * SP * DM * 2);
    rna_f[i] = (float*)alloc((size_t)BSZ * SR * DM * 4);
    rna_b[i] = (u16*)  alloc((size_t)BSZ * SR * DM * 2);
  }
  u16*   qp     = (u16*)  alloc((size_t)maxTok * DM * 2);
  u16*   kp     = (u16*)  alloc((size_t)maxTok * DM * 2);
  u16*   vp     = (u16*)  alloc((size_t)maxTok * DM * 2);
  float* logits = (float*)alloc((size_t)BSZ * HEADS * SP * SP * 4);
  u16*   attnb  = (u16*)  alloc((size_t)BSZ * HEADS * SP * SP * 2);
  u16*   aob    = (u16*)  alloc((size_t)maxTok * DM * 2);
  float* a_f    = (float*)alloc((size_t)maxTok * DM * 4);
  float* o1_f   = (float*)alloc((size_t)maxTok * DM * 4);
  u16*   o1_b   = (u16*)  alloc((size_t)maxTok * DM * 2);
  u16*   mid_b  = (u16*)  alloc((size_t)maxTok * DFF * 2);
  float* f_f    = (float*)alloc((size_t)maxTok * DM * 4);
  float* cnt    = (float*)alloc(256);
  float* xmean  = (float*)alloc((size_t)BSZ * DM * 4);

  const float* at_p  = F(I_AUG_AT);
  const float* aug_p = F(I_AUG_AUG);

  // --- generic GEMM launcher (TB=1 carries the fused logits epilogue) ---
  auto gemm = [&](int TA, int TB, int ACT, const u16* A, const u16* Bm,
                  float* Cf, u16* Cb, const float* bias,
                  int M, int N, int K, int lda, int ldb, int ldc,
                  long sAb, long sAh, long sBb, long sBh, long sCb, long sCh,
                  int H, int batch,
                  const float* ehb, const float* epi, const float* emask, int etransp) {
    dim3 g((unsigned)((N + 63) / 64), (unsigned)((M + 127) / 128), (unsigned)batch);
    dim3 blk(256);
    if (!TA && !TB && !ACT)
      gemm_bf16_wmma<0, 0, 0, 0><<<g, blk, 0, stream>>>(
          A, Bm, Cf, Cb, bias, M, N, K, lda, ldb, ldc,
          sAb, sAh, sBb, sBh, sCb, sCh, H, nullptr, nullptr, nullptr, nullptr, nullptr, 0);
    else if (!TA && TB)
      gemm_bf16_wmma<0, 1, 0, 1><<<g, blk, 0, stream>>>(
          A, Bm, Cf, Cb, bias, M, N, K, lda, ldb, ldc,
          sAb, sAh, sBb, sBh, sCb, sCh, H, ehb, epi, emask, at_p, aug_p, etransp);
    else if (TA && !TB)
      gemm_bf16_wmma<1, 0, 0, 0><<<g, blk, 0, stream>>>(
          A, Bm, Cf, Cb, bias, M, N, K, lda, ldb, ldc,
          sAb, sAh, sBb, sBh, sCb, sCh, H, nullptr, nullptr, nullptr, nullptr, nullptr, 0);
    else
      gemm_bf16_wmma<0, 0, 1, 0><<<g, blk, 0, stream>>>(
          A, Bm, Cf, Cb, bias, M, N, K, lda, ldb, ldc,
          sAb, sAh, sBb, sBh, sCb, sCh, H, nullptr, nullptr, nullptr, nullptr, nullptr, 0);
  };

  // --- one attention + FFN layer ---
  auto attn_layer = [&](const float* q_f, const u16* q_bf, const u16* kv_bf,
                        int Sq, int Sk, const AttnP& P,
                        const float* mask, const float* hb, const float* pi, int transp,
                        const float* rm_ptr, float rm_const,
                        float* out_f, u16* out_bf) {
    const int Mq = BSZ * Sq, Mk = BSZ * Sk;
    // Q/K/V projections (bf16 outputs, layout [B,S,H*D])
    gemm(0, 0, 0, q_bf,  P.wq, nullptr, qp, P.bq, Mq, DM, DM, DM, DM, DM,
         0, 0, 0, 0, 0, 0, 1, 1, nullptr, nullptr, nullptr, 0);
    gemm(0, 0, 0, kv_bf, P.wk, nullptr, kp, P.bk, Mk, DM, DM, DM, DM, DM,
         0, 0, 0, 0, 0, 0, 1, 1, nullptr, nullptr, nullptr, 0);
    gemm(0, 0, 0, kv_bf, P.wv, nullptr, vp, P.bv, Mk, DM, DM, DM, DM, DM,
         0, 0, 0, 0, 0, 0, 1, 1, nullptr, nullptr, nullptr, 0);
    // logits[b,h,k,q] = K . Q^T  with fused aug/mask/scale epilogue
    gemm(0, 1, 0, kp, qp, logits, nullptr, nullptr, Sk, Sq, DEPTH, DM, DM, Sq,
         (long)Sk * DM, DEPTH, (long)Sq * DM, DEPTH,
         (long)HEADS * Sk * Sq, (long)Sk * Sq, HEADS, BSZ * HEADS,
         hb, pi, mask, transp);
    // joint-2D softmax + row_mult, bf16 attn out
    softmax2d_kernel<<<dim3(BSZ * HEADS), dim3(256), 0, stream>>>(
        logits, attnb, (long)Sk * Sq, rm_ptr, rm_const);
    // out[b,h,q,d] = attn^T . V  -> written at [B,Sq,H*D]
    gemm(1, 0, 0, attnb, vp, nullptr, aob, nullptr, Sq, DEPTH, Sk, Sq, DM, DM,
         (long)HEADS * Sk * Sq, (long)Sk * Sq, (long)Sk * DM, DEPTH,
         (long)Sq * DM, DEPTH, HEADS, BSZ * HEADS, nullptr, nullptr, nullptr, 0);
    // output projection
    gemm(0, 0, 0, aob, P.wo, a_f, nullptr, P.bo, Mq, DM, DM, DM, DM, DM,
         0, 0, 0, 0, 0, 0, 1, 1, nullptr, nullptr, nullptr, 0);
    // out1 = LN(q + attn)
    add_ln_kernel<<<dim3((unsigned)Mq), dim3(256), 0, stream>>>(q_f, a_f, P.g1, P.b1, o1_f, o1_b);
    // FFN (ReLU fused into first GEMM epilogue)
    gemm(0, 0, 1, o1_b, P.f1, nullptr, mid_b, P.fb1, Mq, P.dff, DM, DM, P.dff, P.dff,
         0, 0, 0, 0, 0, 0, 1, 1, nullptr, nullptr, nullptr, 0);
    gemm(0, 0, 0, mid_b, P.f2, f_f, nullptr, P.fb2, Mq, DM, P.dff, P.dff, DM, DM,
         0, 0, 0, 0, 0, 0, 1, 1, nullptr, nullptr, nullptr, 0);
    // out = LN(out1 + ffn)
    add_ln_kernel<<<dim3((unsigned)Mq), dim3(256), 0, stream>>>(o1_f, f_f, P.g2, P.b2, out_f, out_bf);
  };

  // --- embeddings + pro_cnt ---
  count_nz_kernel<<<dim3(1), dim3(256), 0, stream>>>((const int*)d_in[I_PROTOK], cnt, SP);
  {
    long tp = (long)BSZ * SP * DM;
    embed_kernel<<<dim3((unsigned)((tp + 255) / 256)), dim3(256), 0, stream>>>(
        (const int*)d_in[I_PROTOK], F(I_PRO_EMB), pro_f[0], pro_b[0], SP, tp);
    long tr = (long)BSZ * SR * DM;
    embed_kernel<<<dim3((unsigned)((tr + 255) / 256)), dim3(256), 0, stream>>>(
        (const int*)d_in[I_RNATOK], F(I_RNA_EMB), rna_f[0], rna_b[0], SR, tr);
  }

  const float* selfMask  = F(I_SELF_MASK);
  const float* crossMask = F(I_CROSS_MASK);
  const float* hb = F(I_HB);
  const float* pi = F(I_PI);

  int pc = 0, rc = 0;
  // protein self-attention stack (shared weights, 2 iterations)
  for (int it = 0; it < 2; ++it) {
    attn_layer(pro_f[pc], pro_b[pc], pro_b[pc], SP, SP, pro_self,
               selfMask, nullptr, nullptr, 0, cnt, 0.f, pro_f[1 - pc], pro_b[1 - pc]);
    pc ^= 1;
  }
  // RNA self-attention stack
  for (int it = 0; it < 2; ++it) {
    attn_layer(rna_f[rc], rna_b[rc], rna_b[rc], SR, SR, rna_self,
               nullptr, nullptr, nullptr, 0, nullptr, (float)SR, rna_f[1 - rc], rna_b[1 - rc]);
    rc ^= 1;
  }
  // cross-attention iterations (pro_new must see the OLD rna)
  for (int it = 0; it < 2; ++it) {
    // rna_new = layer(q=rna, kv=pro): logits [B,H,SP,SR], natural tab/mask, row_mult=pro_cnt
    attn_layer(rna_f[rc], rna_b[rc], pro_b[pc], SR, SP, rna_cross,
               crossMask, hb, pi, 0, cnt, 0.f, rna_f[1 - rc], rna_b[1 - rc]);
    // pro_new = layer(q=pro, kv=rna): logits [B,H,SR,SP], transposed tab/mask, row_mult=SR
    attn_layer(pro_f[pc], pro_b[pc], rna_b[rc], SP, SR, pro_cross,
               crossMask, hb, pi, 1, nullptr, (float)SR, pro_f[1 - pc], pro_b[1 - pc]);
    rc ^= 1; pc ^= 1;
  }

  // --- head ---
  meanpool_kernel<<<dim3((BSZ * DM + 255) / 256), dim3(256), 0, stream>>>(
      pro_f[pc], rna_f[rc], xmean);
  head_kernel<<<dim3(BSZ), dim3(128), 0, stream>>>(
      xmean, F(I_FIN_W), F(I_FIN_B), (float*)d_out);
}
